// MultiScalePatchTokenizer_11407433138452
// MI455X (gfx1250) — matrix operand order repaired and checked by reading, not compiled, not run on hardware
//
#include <hip/hip_runtime.h>
#include <math.h>

#define BATCH  8
#define NPTS   8192
#define CFEAT  64
#define NPATCH 512
#define NCTR   (BATCH*NPATCH)   // 4096

typedef _Float16 v16h __attribute__((ext_vector_type(16)));
typedef _Float16 v2h  __attribute__((ext_vector_type(2)));
typedef float    v8f  __attribute__((ext_vector_type(8)));

// ---------------------------------------------------------------------------
// Weight packing: f32 row-major [K_true][256] -> f16 WMMA-B-fragment-major.
// dst[((kc*16 + nt)*32 + lane)*16 + i] where lane = kh*16 + n, i = 2*v + lo,
// element = w[kc*32 + (v>>2)*16 + kh*8 + (v&3)*2 + lo][nt*16 + n]  (zero-padded K rows).
// One 16KB chunk per kc; each lane's fragment is 32 contiguous bytes.
// ---------------------------------------------------------------------------
__global__ void __launch_bounds__(256)
pack_wfrag(const float* __restrict__ src, _Float16* __restrict__ dst,
           int rowsTrue, int KC) {
  int idx = blockIdx.x * 256 + threadIdx.x;
  int total = KC * 8192;                 // KC * 16 nt * 32 lanes * 16 halfs
  if (idx >= total) return;
  int frag = idx >> 9, r = idx & 511;
  int lane = r >> 4, i = r & 15;
  int kc = frag >> 4, nt = frag & 15;
  int kh = lane >> 4, n = lane & 15;
  int v = i >> 1, lo = i & 1;
  int k = kc * 32 + (v >> 2) * 16 + kh * 8 + (v & 3) * 2 + lo;
  int col = nt * 16 + n;
  dst[idx] = (k < rowsTrue) ? (_Float16)src[k * 256 + col] : (_Float16)0.f;
}

// ---------------------------------------------------------------------------
// Farthest point sampling: one block per batch. Points AND running min-dist
// live in registers (8 points/thread); only the per-iteration argmax reduce
// and the 3-float centroid broadcast touch LDS. Tie-break = lowest index,
// matching jnp.argmax.
// ---------------------------------------------------------------------------
__global__ void __launch_bounds__(1024)
fps_kernel(const float* __restrict__ xyz, int* __restrict__ cidx) {
  __shared__ float wval[32];
  __shared__ int   widx[32];
  __shared__ float ctr[3];
  __shared__ int   far_sh;
  int b = blockIdx.x, tid = threadIdx.x;
  const float* X = xyz + (size_t)b * NPTS * 3;
  float px[8], py[8], pz[8], dist[8];
#pragma unroll
  for (int j = 0; j < 8; ++j) {
    int p = tid + (j << 10);
    px[j] = X[p * 3 + 0];
    py[j] = X[p * 3 + 1];
    pz[j] = X[p * 3 + 2];
    dist[j] = 1e10f;
  }
  if (tid == 0) { ctr[0] = px[0]; ctr[1] = py[0]; ctr[2] = pz[0]; }
  __syncthreads();
  int far = 0;
  for (int it = 0; it < NPATCH; ++it) {
    if (tid == 0) cidx[b * NPATCH + it] = far;
    float cx = ctr[0], cy = ctr[1], cz = ctr[2];
    float bv = -1.f; int bi = 0;
#pragma unroll
    for (int j = 0; j < 8; ++j) {          // ascending p within thread -> strict >
      float dx = px[j] - cx, dy = py[j] - cy, dz = pz[j] - cz;
      float d = dx * dx + dy * dy + dz * dz;
      float nd = fminf(dist[j], d);
      dist[j] = nd;
      if (nd > bv) { bv = nd; bi = tid + (j << 10); }
    }
#pragma unroll
    for (int o = 16; o; o >>= 1) {
      float ov = __shfl_down(bv, o, 32);
      int   oi = __shfl_down(bi, o, 32);
      if (ov > bv || (ov == bv && oi < bi)) { bv = ov; bi = oi; }
    }
    if ((tid & 31) == 0) { wval[tid >> 5] = bv; widx[tid >> 5] = bi; }
    __syncthreads();
    if (tid < 32) {
      bv = wval[tid]; bi = widx[tid];
#pragma unroll
      for (int o = 16; o; o >>= 1) {
        float ov = __shfl_down(bv, o, 32);
        int   oi = __shfl_down(bi, o, 32);
        if (ov > bv || (ov == bv && oi < bi)) { bv = ov; bi = oi; }
      }
      if (tid == 0) far_sh = bi;
    }
    __syncthreads();
    far = far_sh;
    int oj = far >> 10;
    if (tid == (far & 1023)) {
#pragma unroll
      for (int j = 0; j < 8; ++j)
        if (j == oj) { ctr[0] = px[j]; ctr[1] = py[j]; ctr[2] = pz[j]; }
    }
    __syncthreads();
  }
}

__global__ void __launch_bounds__(256)
gather_centers_kernel(const float* __restrict__ xyz, const int* __restrict__ cidx,
                      float* __restrict__ pc) {
  int gc = blockIdx.x * 256 + threadIdx.x;
  if (gc >= NCTR) return;
  int b = gc >> 9;
  int p = cidx[gc];
  pc[gc * 3 + 0] = xyz[(b * NPTS + p) * 3 + 0];
  pc[gc * 3 + 1] = xyz[(b * NPTS + p) * 3 + 1];
  pc[gc * 3 + 2] = xyz[(b * NPTS + p) * 3 + 2];
}

// ---------------------------------------------------------------------------
// Ball query: one wave per center; ordered stream-compaction of the first K
// in-radius point indices via wave32 ballot; pad tail with first hit.
// ---------------------------------------------------------------------------
__global__ void __launch_bounds__(32)
ball_query_kernel(const float* __restrict__ xyz, const float* __restrict__ pc,
                  int* __restrict__ n0, int* __restrict__ n1, int* __restrict__ n2) {
  int gc = blockIdx.x, b = gc >> 9;
  int lane = threadIdx.x;
  float cx = pc[gc * 3], cy = pc[gc * 3 + 1], cz = pc[gc * 3 + 2];
  const float r2[3] = {0.01f, 0.04f, 0.16f};
  const int   Kk[3] = {16, 32, 64};
  int* outp[3] = { n0 + gc * 16, n1 + gc * 32, n2 + gc * 64 };
  int cnt[3] = {0, 0, 0};
  int first[3] = {0, 0, 0};
  unsigned ltm = (1u << lane) - 1u;
  const float* X = xyz + (size_t)b * NPTS * 3;
  for (int basep = 0; basep < NPTS; basep += 32) {
    if (cnt[0] >= 16 && cnt[1] >= 32 && cnt[2] >= 64) break;
    int p = basep + lane;
    float dx = X[p * 3] - cx, dy = X[p * 3 + 1] - cy, dz = X[p * 3 + 2] - cz;
    float d = dx * dx + dy * dy + dz * dz;
#pragma unroll
    for (int s = 0; s < 3; ++s) {
      if (cnt[s] >= Kk[s]) continue;            // uniform branch (cnt from ballot)
      bool in = d <= r2[s];
      unsigned mask = (unsigned)__ballot(in);
      if (cnt[s] == 0 && mask) first[s] = basep + __builtin_ctz(mask);
      int rank = __popc(mask & ltm);
      if (in && (cnt[s] + rank) < Kk[s]) outp[s][cnt[s] + rank] = p;
      int nc = cnt[s] + __popc(mask);
      cnt[s] = nc > Kk[s] ? Kk[s] : nc;
    }
  }
#pragma unroll
  for (int s = 0; s < 3; ++s)
    for (int i = cnt[s] + lane; i < Kk[s]; i += 32) outp[s][i] = first[s];
}

// ---------------------------------------------------------------------------
// WMMA helpers.
// ---------------------------------------------------------------------------
__device__ __forceinline__ void zero_acc(v8f* acc, int nacc) {
  for (int t = 0; t < nacc; ++t)
#pragma unroll
    for (int r = 0; r < 8; ++r) acc[t][r] = 0.f;
}

__device__ __forceinline__ v16h load_a_frag(const _Float16* rp, int half) {
  v16h a;
#pragma unroll
  for (int v = 0; v < 8; ++v) {
    int k = (v >> 2) * 16 + half * 8 + (v & 3) * 2;
    v2h pr = *(const v2h*)(rp + k);              // dword LDS load (pairs contiguous)
    a[2 * v] = pr[0]; a[2 * v + 1] = pr[1];
  }
  return a;
}

// Block-cooperative GEMM for 128-thread blocks (4 waves x 16-row strips,
// each wave owns all 16 N-tiles of its strip). Weights stream through a 16KB
// LDS chunk buffer shared by all 4 waves: while chunk kc feeds 16 WMMAs per
// wave from LDS (ds_load_b128), the global loads for chunk kc+1 sit in
// registers in flight -> L2 latency hidden behind matrix work.
template<int KC, int LDA>
__device__ __forceinline__ void gemm_block128(const _Float16* A, int row0,
                                              const _Float16* __restrict__ Wpk,
                                              _Float16* Wbuf, v8f acc[16],
                                              int tid) {
  int lane = tid & 31;
  int m = lane & 15, half = lane >> 4;
  const int4* gsrc = (const int4*)Wpk;       // 1024 int4 per 16KB chunk
  int4* lbuf = (int4*)Wbuf;
  int4 regs[8];
#pragma unroll
  for (int u = 0; u < 8; ++u) regs[u] = gsrc[u * 128 + tid];        // chunk 0
#pragma unroll
  for (int u = 0; u < 8; ++u) lbuf[u * 128 + tid] = regs[u];
  __syncthreads();
  for (int kc = 0; kc < KC; ++kc) {
    if (kc + 1 < KC) {                       // prefetch next chunk into regs
#pragma unroll
      for (int u = 0; u < 8; ++u) regs[u] = gsrc[(kc + 1) * 1024 + u * 128 + tid];
      __builtin_amdgcn_sched_barrier(0);     // pin prefetch before the WMMAs
    }
    v16h a = load_a_frag(A + (row0 + m) * LDA + kc * 32, half);
    const _Float16* wl = Wbuf + lane * 16;
#pragma unroll
    for (int nt = 0; nt < 16; ++nt) {
      v16h bf = *(const v16h*)(wl + nt * 512);     // 2x ds_load_b128
      acc[nt] = __builtin_amdgcn_wmma_f32_16x16x32_f16(
          false, a, false, bf, (short)0, acc[nt], false, false);
    }
    if (kc + 1 < KC) {
      __syncthreads();                       // all waves done reading chunk kc
#pragma unroll
      for (int u = 0; u < 8; ++u) lbuf[u * 128 + tid] = regs[u];
      __syncthreads();                       // chunk kc+1 visible to all waves
    }
  }
}

// Row stats fully in registers: row m = half*8 + r lives on the 16 lanes of one
// EXEC half, so reduce with width-16 xor shuffles.
__device__ __forceinline__ void row_stats(const v8f acc[16], const float* __restrict__ bias,
                                          int lane, float mean[8], float rstd[8]) {
  int n = lane & 15;
  float s[8], q[8];
#pragma unroll
  for (int r = 0; r < 8; ++r) { s[r] = 0.f; q[r] = 0.f; }
#pragma unroll
  for (int nt = 0; nt < 16; ++nt) {
    float bb = bias[nt * 16 + n];
#pragma unroll
    for (int r = 0; r < 8; ++r) { float x = acc[nt][r] + bb; s[r] += x; q[r] += x * x; }
  }
#pragma unroll
  for (int r = 0; r < 8; ++r) {
#pragma unroll
    for (int off = 8; off; off >>= 1) {
      s[r] += __shfl_xor(s[r], off, 16);
      q[r] += __shfl_xor(q[r], off, 16);
    }
    float m = s[r] * (1.f / 256.f);
    float var = q[r] * (1.f / 256.f) - m * m;
    mean[r] = m; rstd[r] = rsqrtf(var + 1e-5f);
  }
}

template<bool GELU>
__device__ __forceinline__ void ln_to_lds(const v8f acc[16], const float* __restrict__ bias,
                                          const float* __restrict__ g, const float* __restrict__ beta,
                                          int lane, _Float16* __restrict__ out, int row0) {
  float mean[8], rstd[8];
  row_stats(acc, bias, lane, mean, rstd);
  int n = lane & 15, half = lane >> 4;
#pragma unroll
  for (int nt = 0; nt < 16; ++nt) {
    int col = nt * 16 + n;
    float bb = bias[col], gg = g[col], be = beta[col];
#pragma unroll
    for (int r = 0; r < 8; ++r) {
      float y = (acc[nt][r] + bb - mean[r]) * rstd[r] * gg + be;
      if (GELU) y = 0.5f * y * (1.f + erff(y * 0.70710678118654752f));
      out[(row0 + half * 8 + r) * 256 + col] = (_Float16)y;
    }
  }
}

// ---------------------------------------------------------------------------
// Per-scale grouped MLP, 128 threads / 64 rows per block (64/K centers; one
// wave per 16-row strip). Stage [64 x 96] (feat|rel|sin-cos PE, zero-padded
// 91->96) -> GEMM(96->256) -> LN+GELU -> GEMM(256->256) -> LN -> per-center
// column max over K. LDS: A1 12KB + A2 32KB + Wbuf 16KB = 60KB.
// waves_per_eu(2,8): occupancy is LDS-bound; let the allocator use enough
// VGPRs for acc[16] + prefetch regs (~210) without scratch spills.
// ---------------------------------------------------------------------------
template<int K>
__global__ void __launch_bounds__(128) __attribute__((amdgpu_waves_per_eu(2, 8)))
scale_kernel(const float* __restrict__ xyz, const float* __restrict__ feat,
             const float* __restrict__ pc, const int* __restrict__ nidx,
             const _Float16* __restrict__ w1pk, const float* __restrict__ b1,
             const float* __restrict__ g1, const float* __restrict__ bt1,
             const _Float16* __restrict__ w2pk, const float* __restrict__ b2,
             const float* __restrict__ g2, const float* __restrict__ bt2,
             _Float16* __restrict__ sout) {
  constexpr int CPB = 64 / K;                  // centers per block
  __shared__ _Float16 A1[64][96];
  __shared__ _Float16 A2[64][256];
  __shared__ __align__(16) _Float16 Wbuf[8192];
  int gc0 = blockIdx.x * CPB;
  int tid = threadIdx.x;
  int lane = tid & 31;
  int wv = tid >> 5;

  for (int i = tid; i < 64 * 96; i += 128) {
    int row = i / 96, c = i - row * 96;
    int ci = row / K, j = row - ci * K;
    int gc = gc0 + ci, bb = gc >> 9;
    int p = nidx[gc * K + j];
    float val;
    if (c < 64) {
      val = feat[(bb * NPTS + p) * CFEAT + c];
    } else if (c < 67) {
      int ax = c - 64;
      val = xyz[(bb * NPTS + p) * 3 + ax] - pc[gc * 3 + ax];
    } else if (c < 91) {
      int t = c - 67, ax = t >> 3, jj = t & 7;
      float rel = xyz[(bb * NPTS + p) * 3 + ax] - pc[gc * 3 + ax];
      float fr = __expf(-9.2103403719761836f * (float)(jj & 3) * 0.25f);
      float arg = rel * fr;
      val = (jj < 4) ? __sinf(arg) : __cosf(arg);
    } else {
      val = 0.f;                               // pad 91 -> 96
    }
    A1[row][c] = (_Float16)val;
  }
  __syncthreads();

  v8f acc[16];
  zero_acc(acc, 16);
  gemm_block128<3, 96>(&A1[0][0], wv * 16, w1pk, Wbuf, acc, tid);
  ln_to_lds<true>(acc, b1, g1, bt1, lane, &A2[0][0], wv * 16);
  __syncthreads();

  zero_acc(acc, 16);
  gemm_block128<8, 256>(&A2[0][0], wv * 16, w2pk, Wbuf, acc, tid);
  ln_to_lds<false>(acc, b2, g2, bt2, lane, &A2[0][0], wv * 16);
  __syncthreads();

  for (int i = tid; i < CPB * 256; i += 128) {
    int ci = i >> 8, c = i & 255;
    int r0 = ci * K;
    float mx = (float)A2[r0][c];
#pragma unroll 4
    for (int k = 1; k < K; ++k) mx = fmaxf(mx, (float)A2[r0 + k][c]);
    sout[(size_t)(gc0 + ci) * 256 + c] = (_Float16)mx;
  }
}

// ---------------------------------------------------------------------------
// Projection MLP, 128 threads / 16 centers per block. The 4 waves split the
// 16 N-tiles (4 each, acc = 32 VGPRs); weights stream through the shared
// pipelined LDS chunk buffer; LayerNorm stats combine across waves via a
// small LDS exchange. tok = [feat(64)|3x scale(256)|pos(96)] = 928 dims.
// LDS: tok 29KB + A2 8KB + Wbuf 16KB + stats 0.5KB = 54KB.
// ---------------------------------------------------------------------------
__device__ __forceinline__ void proj_stats(const v8f acc[4], const float* __restrict__ bias,
                                           int tid, float (*st)[2][8][2],
                                           float mean[8], float rstd[8]) {
  int lane = tid & 31, wv = tid >> 5;
  int n = lane & 15, half = lane >> 4;
  float s[8], q[8];
#pragma unroll
  for (int r = 0; r < 8; ++r) { s[r] = 0.f; q[r] = 0.f; }
#pragma unroll
  for (int t = 0; t < 4; ++t) {
    float bb = bias[(wv * 4 + t) * 16 + n];
#pragma unroll
    for (int r = 0; r < 8; ++r) { float x = acc[t][r] + bb; s[r] += x; q[r] += x * x; }
  }
#pragma unroll
  for (int r = 0; r < 8; ++r) {
#pragma unroll
    for (int off = 8; off; off >>= 1) {
      s[r] += __shfl_xor(s[r], off, 16);
      q[r] += __shfl_xor(q[r], off, 16);
    }
  }
  if (n == 0) {
#pragma unroll
    for (int r = 0; r < 8; ++r) { st[wv][half][r][0] = s[r]; st[wv][half][r][1] = q[r]; }
  }
  __syncthreads();
#pragma unroll
  for (int r = 0; r < 8; ++r) {
    float S = 0.f, Q = 0.f;
#pragma unroll
    for (int w2 = 0; w2 < 4; ++w2) { S += st[w2][half][r][0]; Q += st[w2][half][r][1]; }
    float m = S * (1.f / 256.f);
    float var = Q * (1.f / 256.f) - m * m;
    mean[r] = m; rstd[r] = rsqrtf(var + 1e-5f);
  }
  __syncthreads();                             // stats LDS reusable afterwards
}

// Two-pass 4-register staging with scheduling fences so the passes cannot be
// re-merged by the scheduler (which previously recreated the 8-reg live range
// and spilled to scratch).
template<int KC, int LDA>
__device__ __forceinline__ void gemm_proj128(const _Float16* A,
                                             const _Float16* __restrict__ Wpk,
                                             _Float16* Wbuf, v8f acc[4], int tid) {
  int lane = tid & 31, wv = tid >> 5;
  int m = lane & 15, half = lane >> 4;
  const int4* gsrc = (const int4*)Wpk;
  int4* lbuf = (int4*)Wbuf;
  int4 regs[4];
#pragma unroll
  for (int h = 0; h < 2; ++h) {
#pragma unroll
    for (int u = 0; u < 4; ++u) regs[u] = gsrc[(h * 4 + u) * 128 + tid];
#pragma unroll
    for (int u = 0; u < 4; ++u) lbuf[(h * 4 + u) * 128 + tid] = regs[u];
    __builtin_amdgcn_sched_barrier(0);
  }
  __syncthreads();
  for (int kc = 0; kc < KC; ++kc) {
    if (kc + 1 < KC) {                       // prefetch first half of next chunk
#pragma unroll
      for (int u = 0; u < 4; ++u) regs[u] = gsrc[(kc + 1) * 1024 + u * 128 + tid];
      __builtin_amdgcn_sched_barrier(0);     // pin prefetch before the WMMAs
    }
    v16h a = load_a_frag(A + m * LDA + kc * 32, half);
#pragma unroll
    for (int t = 0; t < 4; ++t) {
      v16h bf = *(const v16h*)(Wbuf + ((wv * 4 + t) * 32 + lane) * 16);
      acc[t] = __builtin_amdgcn_wmma_f32_16x16x32_f16(
          false, a, false, bf, (short)0, acc[t], false, false);
    }
    if (kc + 1 < KC) {
      __syncthreads();
#pragma unroll
      for (int u = 0; u < 4; ++u) lbuf[u * 128 + tid] = regs[u];
      __builtin_amdgcn_sched_barrier(0);     // keep the two passes separate
#pragma unroll
      for (int u = 0; u < 4; ++u) regs[u] = gsrc[(kc + 1) * 1024 + (4 + u) * 128 + tid];
#pragma unroll
      for (int u = 0; u < 4; ++u) lbuf[(4 + u) * 128 + tid] = regs[u];
      __syncthreads();
    }
  }
}

__global__ void __launch_bounds__(128) __attribute__((amdgpu_waves_per_eu(2, 8)))
proj_kernel(const float* __restrict__ feat, const float* __restrict__ pc,
            const int* __restrict__ cidx,
            const _Float16* __restrict__ sf0, const _Float16* __restrict__ sf1,
            const _Float16* __restrict__ sf2,
            const _Float16* __restrict__ w1pk, const float* __restrict__ b1,
            const float* __restrict__ g1, const float* __restrict__ bt1,
            const _Float16* __restrict__ w2pk, const float* __restrict__ b2,
            const float* __restrict__ g2, const float* __restrict__ bt2,
            float* __restrict__ out) {
  __shared__ _Float16 tok[16][928];
  __shared__ _Float16 A2[16][256];
  __shared__ __align__(16) _Float16 Wbuf[8192];
  __shared__ float stats[4][2][8][2];
  int base = blockIdx.x * 16;
  int tid = threadIdx.x, lane = tid & 31, wv = tid >> 5;
  int n = lane & 15, half = lane >> 4;

  for (int i = tid; i < 16 * 928; i += 128) {
    int r = i / 928, c = i - r * 928;
    int gc = base + r, b = gc >> 9;
    _Float16 hv;
    if (c < 64) {
      int p = cidx[gc];
      hv = (_Float16)feat[(b * NPTS + p) * CFEAT + c];
    } else if (c < 832) {
      int s = (c - 64) >> 8, cc = (c - 64) & 255;
      const _Float16* sp = (s == 0) ? sf0 : ((s == 1) ? sf1 : sf2);
      hv = sp[(size_t)gc * 256 + cc];
    } else {
      int t = c - 832, ax = t >> 5, j = t & 31, fi = j & 15;
      float x = pc[gc * 3 + ax];
      float fr = __expf(-9.2103403719761836f * (float)fi * (1.f / 16.f));
      float arg = x * fr;
      hv = (_Float16)((j < 16) ? __sinf(arg) : __cosf(arg));
    }
    tok[r][c] = hv;
  }
  __syncthreads();

  v8f acc[4];
  zero_acc(acc, 4);
  gemm_proj128<29, 928>(&tok[0][0], w1pk, Wbuf, acc, tid);
  {
    float mean[8], rstd[8];
    proj_stats(acc, b1, tid, stats, mean, rstd);
#pragma unroll
    for (int t = 0; t < 4; ++t) {
      int col = (wv * 4 + t) * 16 + n;
      float bb = b1[col], gg = g1[col], be = bt1[col];
#pragma unroll
      for (int r = 0; r < 8; ++r) {
        float y = (acc[t][r] + bb - mean[r]) * rstd[r] * gg + be;
        y = 0.5f * y * (1.f + erff(y * 0.70710678118654752f));
        A2[half * 8 + r][col] = (_Float16)y;
      }
    }
  }
  __syncthreads();                             // A2 complete across waves

  zero_acc(acc, 4);
  gemm_proj128<8, 256>(&A2[0][0], w2pk, Wbuf, acc, tid);
  {
    float mean[8], rstd[8];
    proj_stats(acc, b2, tid, stats, mean, rstd);
#pragma unroll
    for (int t = 0; t < 4; ++t) {
      int col = (wv * 4 + t) * 16 + n;
      float bb = b2[col], gg = g2[col], be = bt2[col];
#pragma unroll
      for (int r = 0; r < 8; ++r) {
        float y = (acc[t][r] + bb - mean[r]) * rstd[r] * gg + be;
        out[(size_t)(base + half * 8 + r) * 256 + col] = y;
      }
    }
  }
}

// ---------------------------------------------------------------------------
// Host launcher. Param pointers assume JAX pytree flattening (dict keys
// sorted): [xyz, point_feature, proj.m1.{b,beta,g,w}, proj.m2.{b,beta,g,w},
// scales[0..2].m1.{b,beta,g,w}, .m2.{b,beta,g,w}].
// ---------------------------------------------------------------------------
extern "C" void kernel_launch(void* const* d_in, const int* in_sizes, int n_in,
                              void* d_out, int out_size, void* d_ws, size_t ws_size,
                              hipStream_t stream) {
  (void)in_sizes; (void)n_in; (void)out_size; (void)ws_size;
  const float* xyz  = (const float*)d_in[0];
  const float* feat = (const float*)d_in[1];
  const float* pj1_b  = (const float*)d_in[2];
  const float* pj1_be = (const float*)d_in[3];
  const float* pj1_g  = (const float*)d_in[4];
  const float* pj1_w  = (const float*)d_in[5];
  const float* pj2_b  = (const float*)d_in[6];
  const float* pj2_be = (const float*)d_in[7];
  const float* pj2_g  = (const float*)d_in[8];
  const float* pj2_w  = (const float*)d_in[9];
  const float *s1_b[3], *s1_be[3], *s1_g[3], *s1_w[3];
  const float *s2_b[3], *s2_be[3], *s2_g[3], *s2_w[3];
  for (int s = 0; s < 3; ++s) {
    int o = 10 + s * 8;
    s1_b[s]  = (const float*)d_in[o + 0];
    s1_be[s] = (const float*)d_in[o + 1];
    s1_g[s]  = (const float*)d_in[o + 2];
    s1_w[s]  = (const float*)d_in[o + 3];
    s2_b[s]  = (const float*)d_in[o + 4];
    s2_be[s] = (const float*)d_in[o + 5];
    s2_g[s]  = (const float*)d_in[o + 6];
    s2_w[s]  = (const float*)d_in[o + 7];
  }
  float* out = (float*)d_out;

  char* wsb = (char*)d_ws;
  size_t off = 0;
  auto carve = [&](size_t bytes) -> void* {
    void* p = wsb + off;
    off = (off + bytes + 255) & ~(size_t)255;
    return p;
  };
  int*   cidx = (int*)  carve((size_t)NCTR * sizeof(int));
  float* pc   = (float*)carve((size_t)NCTR * 3 * sizeof(float));
  int* n0 = (int*)carve((size_t)NCTR * 16 * sizeof(int));
  int* n1 = (int*)carve((size_t)NCTR * 32 * sizeof(int));
  int* n2 = (int*)carve((size_t)NCTR * 64 * sizeof(int));
  _Float16 *w1s[3], *w2s[3];
  for (int s = 0; s < 3; ++s) {
    w1s[s] = (_Float16*)carve((size_t)3 * 8192 * 2);
    w2s[s] = (_Float16*)carve((size_t)8 * 8192 * 2);
  }
  _Float16* w1p = (_Float16*)carve((size_t)29 * 8192 * 2);
  _Float16* w2p = (_Float16*)carve((size_t)8 * 8192 * 2);
  _Float16* sf[3];
  for (int s = 0; s < 3; ++s) sf[s] = (_Float16*)carve((size_t)NCTR * 256 * 2);

  for (int s = 0; s < 3; ++s) {
    pack_wfrag<<<(3 * 8192 + 255) / 256, 256, 0, stream>>>(s1_w[s], w1s[s], 91, 3);
    pack_wfrag<<<(8 * 8192 + 255) / 256, 256, 0, stream>>>(s2_w[s], w2s[s], 256, 8);
  }
  pack_wfrag<<<(29 * 8192 + 255) / 256, 256, 0, stream>>>(pj1_w, w1p, 928, 29);
  pack_wfrag<<<(8 * 8192 + 255) / 256, 256, 0, stream>>>(pj2_w, w2p, 256, 8);

  fps_kernel<<<BATCH, 1024, 0, stream>>>(xyz, cidx);
  gather_centers_kernel<<<(NCTR + 255) / 256, 256, 0, stream>>>(xyz, cidx, pc);
  ball_query_kernel<<<NCTR, 32, 0, stream>>>(xyz, pc, n0, n1, n2);

  scale_kernel<16><<<NCTR / 4, 128, 0, stream>>>(xyz, feat, pc, n0,
      w1s[0], s1_b[0], s1_g[0], s1_be[0], w2s[0], s2_b[0], s2_g[0], s2_be[0], sf[0]);
  scale_kernel<32><<<NCTR / 2, 128, 0, stream>>>(xyz, feat, pc, n1,
      w1s[1], s1_b[1], s1_g[1], s1_be[1], w2s[1], s2_b[1], s2_g[1], s2_be[1], sf[1]);
  scale_kernel<64><<<NCTR,     128, 0, stream>>>(xyz, feat, pc, n2,
      w1s[2], s1_b[2], s1_g[2], s1_be[2], w2s[2], s2_b[2], s2_g[2], s2_be[2], sf[2]);

  proj_kernel<<<NCTR / 16, 128, 0, stream>>>(feat, pc, cidx, sf[0], sf[1], sf[2],
      w1p, pj1_b, pj1_g, pj1_be, w2p, pj2_b, pj2_g, pj2_be, out);
}
